// ConvMambaBlock_12146167513466
// MI455X (gfx1250) — matrix operand order repaired and checked by reading, not compiled, not run on hardware
//
#include <hip/hip_runtime.h>
#include <stdint.h>

typedef __bf16 bf16;
typedef __attribute__((ext_vector_type(16))) __bf16 v16bf;
typedef __attribute__((ext_vector_type(8)))  __bf16 v8bf;
typedef __attribute__((ext_vector_type(8)))  float  v8f;
typedef __attribute__((ext_vector_type(4)))  int    v4i;

#define CDIV(a, b) (((a) + (b) - 1) / (b))

#if defined(__HIP_DEVICE_COMPILE__) && defined(__has_builtin)
#  if __has_builtin(__builtin_amdgcn_global_load_async_to_lds_b128) && \
      __has_builtin(__builtin_amdgcn_s_wait_asynccnt)
#    define USE_ASYNC_LDS 1
#  endif
#endif
#ifndef USE_ASYNC_LDS
#  define USE_ASYNC_LDS 0
#endif

#if USE_ASYNC_LDS
typedef __attribute__((address_space(1))) v4i as1_v4i;   // global int4
typedef __attribute__((address_space(3))) v4i as3_v4i;   // LDS int4
#endif

// ---------------------------------------------------------------------------
// Register-blocked WMMA bf16 GEMM with LDS-staged weights.
//   C[M,N] = A[M,K] * Bpack (+ bias), f32 accumulate.
// Grid: (M/128, N/64); block = 256 threads (8 waves).
// Each wave computes one 16-row M tile x 64 columns (4 16x16 WMMA tiles):
// one A-fragment load feeds 4 WMMAs; B fragments come from LDS.
// Bpack layout (fragment-native): chunk f = (nt*KB + kb)*32 + lane is a v16bf
// where element i is  k = kb*32 + 8*(lane>>4) + i (i<8) / kb*32 + 16 + 8*(lane>>4) + (i-8) (i>=8),  n = nt*16 + (lane&15).
// The 64-column slab [nt0, nt0+4) is a single contiguous region -> staged to
// LDS with GLOBAL_LOAD_ASYNC_TO_LDS_B128 (ASYNCcnt) when available.
// ---------------------------------------------------------------------------
__global__ __launch_bounds__(256) void gemm_wmma_bf16_blk(
    const bf16* __restrict__ A, const bf16* __restrict__ Bp,
    const float* __restrict__ bias, float* __restrict__ C,
    int M, int N, int K)
{
    extern __shared__ __align__(32) char smem_raw[];
    bf16* sB = (bf16*)smem_raw;                     // 4 * KB * 512 bf16

    const int KB   = K >> 5;
    const int tid  = threadIdx.x;
    const int wave = tid >> 5;
    const int lane = tid & 31;
    const int half = lane >> 4;
    const int lidx = lane & 15;
    const int m_tile = blockIdx.x * 8 + wave;
    const int nt0    = blockIdx.y * 4;

    // ---- stage the B slab (contiguous) into LDS --------------------------
    const bf16* gB = Bp + (size_t)nt0 * KB * 512;
    const int chunks = KB * 256;                    // 16-byte chunks in slab
#if USE_ASYNC_LDS
    for (int i = tid; i < chunks; i += 256) {
        __builtin_amdgcn_global_load_async_to_lds_b128(
            (as1_v4i*)(gB + (size_t)i * 8),
            (as3_v4i*)(sB + (size_t)i * 8),
            0, 0);
    }
    __builtin_amdgcn_s_wait_asynccnt(0);
#else
    for (int i = tid; i < chunks; i += 256) {
        *(v8bf*)(sB + (size_t)i * 8) = *(const v8bf*)(gB + (size_t)i * 8);
    }
#endif
    __syncthreads();

    // ---- main K loop -----------------------------------------------------
    const bf16* arow = A + (size_t)(m_tile * 16 + lidx) * K;
    const v16bf* sBf = (const v16bf*)sB;

    v8f acc0 = {}, acc1 = {}, acc2 = {}, acc3 = {};
    for (int kb = 0; kb < KB; ++kb) {
        const int k0 = kb << 5;
        v8bf alo = *(const v8bf*)(arow + k0 + 8 * half);
        v8bf ahi = *(const v8bf*)(arow + k0 + 16 + 8 * half);
        v16bf av = __builtin_shufflevector(alo, ahi,
                                           0, 1, 2, 3, 4, 5, 6, 7,
                                           8, 9, 10, 11, 12, 13, 14, 15);
        v16bf b0 = sBf[((0 * KB + kb) << 5) + lane];
        v16bf b1 = sBf[((1 * KB + kb) << 5) + lane];
        v16bf b2 = sBf[((2 * KB + kb) << 5) + lane];
        v16bf b3 = sBf[((3 * KB + kb) << 5) + lane];
        acc0 = __builtin_amdgcn_wmma_f32_16x16x32_bf16(false, av, false, b0, (short)0, acc0, false, false);
        acc1 = __builtin_amdgcn_wmma_f32_16x16x32_bf16(false, av, false, b1, (short)0, acc1, false, false);
        acc2 = __builtin_amdgcn_wmma_f32_16x16x32_bf16(false, av, false, b2, (short)0, acc2, false, false);
        acc3 = __builtin_amdgcn_wmma_f32_16x16x32_bf16(false, av, false, b3, (short)0, acc3, false, false);
    }

    // ---- write out -------------------------------------------------------
    v8f accs[4] = { acc0, acc1, acc2, acc3 };
    #pragma unroll
    for (int j = 0; j < 4; ++j) {
        const int ncol = (nt0 + j) * 16 + lidx;
        const float bb = bias ? bias[ncol] : 0.0f;
        #pragma unroll
        for (int v = 0; v < 8; ++v) {
            const int mrow = m_tile * 16 + v + 8 * half;
            C[(size_t)mrow * N + ncol] = accs[j][v] + bb;
        }
    }
}

// Pack a (N,K) row-major f32 weight (torch (out,in)) into the fragment-native
// bf16 layout consumed by gemm_wmma_bf16_blk (implicit transpose to KxN).
__global__ void pack_b_bf16(const float* __restrict__ W, bf16* __restrict__ out,
                            int N, int K)
{
    const int idx = blockIdx.x * blockDim.x + threadIdx.x;
    if (idx >= N * K) return;
    const int i    = idx & 15;
    int t          = idx >> 4;
    const int lane = t & 31; t >>= 5;
    const int KB   = K >> 5;
    const int kb   = t % KB;
    const int nt   = t / KB;
    const int k    = kb * 32 + ((i < 8) ? 8 * (lane >> 4) + i : 16 + 8 * (lane >> 4) + (i - 8));
    const int n    = nt * 16 + (lane & 15);
    out[idx] = (bf16)W[(size_t)n * K + k];
}

// im2col for the front conv: x (B, Cin, L) f32 -> A (B*L, Cin*Kw) bf16, SAME pad.
__global__ void im2col_conv(const float* __restrict__ x, bf16* __restrict__ A,
                            int B, int Cin, int L, int Kw, int pad)
{
    const long long idx = (long long)blockIdx.x * blockDim.x + threadIdx.x;
    const long long total = (long long)B * L * Cin * Kw;
    if (idx >= total) return;
    const int CK  = Cin * Kw;
    const int col = (int)(idx % CK);
    const long long m = idx / CK;
    const int l = (int)(m % L);
    const int b = (int)(m / L);
    const int c  = col / Kw;
    const int kk = col % Kw;
    const int ls = l + kk - pad;
    float v = (ls >= 0 && ls < L) ? x[((size_t)b * Cin + c) * L + ls] : 0.0f;
    A[idx] = (bf16)v;
}

// LayerNorm over D=128 channels per token; one wave per token.
// out = LN(in + pre_add)*w + b + post_add   (pre_add / post_add optional)
__global__ __launch_bounds__(256) void ln_tokens(
    const float* __restrict__ in, const float* __restrict__ pre_add,
    const float* __restrict__ w, const float* __restrict__ b,
    const float* __restrict__ post_add, float* __restrict__ out, int M)
{
    const int wave = blockIdx.x * (blockDim.x / 32) + (threadIdx.x / 32);
    const int lane = threadIdx.x & 31;
    if (wave >= M) return;
    const size_t base = (size_t)wave * 128;

    float v[4];
    float s = 0.0f;
    #pragma unroll
    for (int i = 0; i < 4; ++i) {
        const int ch = lane + 32 * i;
        float t = in[base + ch];
        if (pre_add) t += pre_add[base + ch];
        v[i] = t;
        s += t;
    }
    #pragma unroll
    for (int off = 16; off >= 1; off >>= 1) s += __shfl_xor(s, off, 32);
    const float mu = s * (1.0f / 128.0f);

    float q = 0.0f;
    #pragma unroll
    for (int i = 0; i < 4; ++i) { const float d = v[i] - mu; q += d * d; }
    #pragma unroll
    for (int off = 16; off >= 1; off >>= 1) q += __shfl_xor(q, off, 32);
    const float rinv = rsqrtf(q * (1.0f / 128.0f) + 1e-5f);

    #pragma unroll
    for (int i = 0; i < 4; ++i) {
        const int ch = lane + 32 * i;
        float o = (v[i] - mu) * rinv * w[ch] + b[ch];
        if (post_add) o += post_add[base + ch];
        out[base + ch] = o;
    }
}

// Depthwise causal conv (d_conv=4) over L + SiLU.  Input = xz[:, :256].
__global__ void dwconv_silu(const float* __restrict__ xz,
                            const float* __restrict__ cw,
                            const float* __restrict__ cb,
                            float* __restrict__ xm, int B, int L)
{
    const long long idx = (long long)blockIdx.x * blockDim.x + threadIdx.x;
    const long long total = (long long)B * L * 256;
    if (idx >= total) return;
    const int d = (int)(idx & 255);
    const long long m = idx >> 8;
    const int l = (int)(m % L);
    const int b = (int)(m / L);
    float acc = cb[d];
    #pragma unroll
    for (int j = 0; j < 4; ++j) {
        const int ls = l - 3 + j;
        if (ls >= 0)
            acc += xz[((size_t)b * L + ls) * 512 + d] * cw[d * 4 + j];
    }
    xm[(size_t)m * 256 + d] = acc / (1.0f + __expf(-acc));
}

// x_proj: xdbl[m, j] = xm[m,:] . w[j,:],  j < 40 (dt_rank=8 + 2*d_state=32)
__global__ void xproj_kernel(const float* __restrict__ xm,
                             const float* __restrict__ w,
                             float* __restrict__ xdbl, int M)
{
    const int idx = blockIdx.x * blockDim.x + threadIdx.x;
    if (idx >= M * 40) return;
    const int j = idx % 40;
    const int m = idx / 40;
    const float* r  = xm + (size_t)m * 256;
    const float* wr = w + (size_t)j * 256;
    float acc = 0.0f;
    #pragma unroll 4
    for (int k = 0; k < 256; ++k) acc += r[k] * wr[k];
    xdbl[(size_t)m * 40 + j] = acc;
}

// dt_proj (rank 8) + softplus -> delta[m, d]
__global__ void dtproj_softplus(const float* __restrict__ xdbl,
                                const float* __restrict__ dtw,
                                const float* __restrict__ dtb,
                                float* __restrict__ delta, int M)
{
    const long long idx = (long long)blockIdx.x * blockDim.x + threadIdx.x;
    if (idx >= (long long)M * 256) return;
    const int d = (int)(idx & 255);
    const long long m = idx >> 8;
    float acc = dtb[d];
    #pragma unroll
    for (int r = 0; r < 8; ++r) acc += xdbl[m * 40 + r] * dtw[d * 8 + r];
    delta[idx] = (acc > 20.0f) ? acc : log1pf(__expf(acc));
}

// Selective scan: one wave handles 2 (batch,channel) pairs; n = lane&15 holds
// one of the 16 SSM states; y_t = sum_n h*C via 16-wide xor-shuffle reduction.
__global__ __launch_bounds__(32) void scan_kernel(
    const float* __restrict__ delta, const float* __restrict__ xdbl,
    const float* __restrict__ xm, const float* __restrict__ A_log,
    float* __restrict__ yv, int L)
{
    const int lane = threadIdx.x;
    const int half = lane >> 4;
    const int n    = lane & 15;
    const int g    = blockIdx.x * 2 + half;   // global channel in [0, B*256)
    const int d    = g & 255;
    const int b    = g >> 8;
    const float Aval = -__expf(A_log[d * 16 + n]);
    float h = 0.0f;
    const size_t base = (size_t)b * L;
    for (int t = 0; t < L; ++t) {
        const size_t m = base + t;
        const float dl = delta[m * 256 + d];
        const float Bv = xdbl[m * 40 + 8 + n];
        const float Cv = xdbl[m * 40 + 24 + n];
        const float xv = xm[m * 256 + d];
        h = __expf(dl * Aval) * h + dl * Bv * xv;
        float p = h * Cv;
        #pragma unroll
        for (int off = 8; off >= 1; off >>= 1) p += __shfl_xor(p, off, 16);
        if (n == 0) yv[m * 256 + d] = p;
    }
}

// y = (y + xm*D) * silu(z)   (z = xz[:, 256:512])
__global__ void ygate_kernel(float* __restrict__ yv, const float* __restrict__ xm,
                             const float* __restrict__ Dp, const float* __restrict__ xz,
                             int M)
{
    const long long idx = (long long)blockIdx.x * blockDim.x + threadIdx.x;
    if (idx >= (long long)M * 256) return;
    const int d = (int)(idx & 255);
    const long long m = idx >> 8;
    const float z  = xz[(size_t)m * 512 + 256 + d];
    const float sz = z / (1.0f + __expf(-z));
    yv[idx] = (yv[idx] + xm[idx] * Dp[d]) * sz;
}

__global__ void cvt_f32_bf16(const float* __restrict__ src, bf16* __restrict__ dst,
                             long long n)
{
    const long long i = (long long)blockIdx.x * blockDim.x + threadIdx.x;
    if (i < n) dst[i] = (bf16)src[i];
}

// token-major (B*L, 128) -> channel-major (B, 128, L) output
__global__ void transpose_out(const float* __restrict__ src, float* __restrict__ dst,
                              int B, int L)
{
    const long long idx = (long long)blockIdx.x * blockDim.x + threadIdx.x;
    if (idx >= (long long)B * L * 128) return;
    const int d = (int)(idx & 127);
    const long long m = idx >> 7;
    const int l = (int)(m % L);
    const int b = (int)(m / L);
    dst[((size_t)b * 128 + d) * L + l] = src[(size_t)m * 128 + d];
}

// ---------------------------------------------------------------------------
// Host orchestration
// ---------------------------------------------------------------------------
struct MambaParams {
    const float *in_proj_w, *conv_w, *conv_b, *x_proj_w;
    const float *dt_proj_w, *dt_proj_b, *A_log, *D, *out_proj_w;
};

static inline void launch_gemm(const bf16* A, const bf16* Bp, const float* bias,
                               float* C, int M, int N, int K, hipStream_t stream)
{
    const size_t smem = (size_t)(K / 32) * 4096;   // 4 n-tiles * KB * 1KB
    gemm_wmma_bf16_blk<<<dim3(M / 128, N / 64), 256, smem, stream>>>(A, Bp, bias, C, M, N, K);
}

static void run_mamba(const MambaParams& p, const float* tok_in, float* tok_out,
                      float* XZ, float* XM, float* XDBL, float* DELTA, float* YV,
                      bf16* ABF, bf16* WPACK, hipStream_t stream)
{
    const int B = 4, L = 4096, M = B * L;
    // in_proj: (M,128) @ (128,512)
    cvt_f32_bf16<<<CDIV((long long)M * 128, 256), 256, 0, stream>>>(tok_in, ABF, (long long)M * 128);
    pack_b_bf16<<<CDIV(512 * 128, 256), 256, 0, stream>>>(p.in_proj_w, WPACK, 512, 128);
    launch_gemm(ABF, WPACK, nullptr, XZ, M, 512, 128, stream);
    // depthwise conv + silu
    dwconv_silu<<<CDIV((long long)M * 256, 256), 256, 0, stream>>>(XZ, p.conv_w, p.conv_b, XM, B, L);
    // x_proj
    xproj_kernel<<<CDIV(M * 40, 256), 256, 0, stream>>>(XM, p.x_proj_w, XDBL, M);
    // dt_proj + softplus
    dtproj_softplus<<<CDIV((long long)M * 256, 256), 256, 0, stream>>>(XDBL, p.dt_proj_w, p.dt_proj_b, DELTA, M);
    // selective scan
    scan_kernel<<<B * 256 / 2, 32, 0, stream>>>(DELTA, XDBL, XM, p.A_log, YV, L);
    // gate
    ygate_kernel<<<CDIV((long long)M * 256, 256), 256, 0, stream>>>(YV, XM, p.D, XZ, M);
    // out_proj: (M,256) @ (256,128)
    cvt_f32_bf16<<<CDIV((long long)M * 256, 256), 256, 0, stream>>>(YV, ABF, (long long)M * 256);
    pack_b_bf16<<<CDIV(128 * 256, 256), 256, 0, stream>>>(p.out_proj_w, WPACK, 128, 256);
    launch_gemm(ABF, WPACK, nullptr, tok_out, M, 128, 256, stream);
}

extern "C" void kernel_launch(void* const* d_in, const int* in_sizes, int n_in,
                              void* d_out, int out_size, void* d_ws, size_t ws_size,
                              hipStream_t stream)
{
    const int B = 4, L = 4096, IN = 64, M = B * L;

    const float* x        = (const float*)d_in[0];
    const float* conv_w   = (const float*)d_in[1];
    const float* conv_b   = (const float*)d_in[2];
    const float* convln_w = (const float*)d_in[3];
    const float* convln_b = (const float*)d_in[4];
    const float* ln1_w    = (const float*)d_in[5];
    const float* ln1_b    = (const float*)d_in[6];
    const float* ln2_w    = (const float*)d_in[7];
    const float* ln2_b    = (const float*)d_in[8];
    MambaParams m1 = { (const float*)d_in[9],  (const float*)d_in[10], (const float*)d_in[11],
                       (const float*)d_in[12], (const float*)d_in[13], (const float*)d_in[14],
                       (const float*)d_in[15], (const float*)d_in[16], (const float*)d_in[17] };
    MambaParams m2 = { (const float*)d_in[18], (const float*)d_in[19], (const float*)d_in[20],
                       (const float*)d_in[21], (const float*)d_in[22], (const float*)d_in[23],
                       (const float*)d_in[24], (const float*)d_in[25], (const float*)d_in[26] };
    float* out = (float*)d_out;

    // workspace arena
    uint8_t* ws = (uint8_t*)d_ws;
    size_t off = 0;
    auto alloc = [&](size_t bytes) -> void* {
        void* p = ws + off;
        off += (bytes + 255) & ~(size_t)255;
        return p;
    };
    float* T0    = (float*)alloc((size_t)M * 128 * 4);   // conv out / res1; later mamba2 out
    float* T1    = (float*)alloc((size_t)M * 128 * 4);   // convln out; later ln1 out, final LN out
    float* T2    = (float*)alloc((size_t)M * 128 * 4);   // mamba1 out / res2
    float* XZ    = (float*)alloc((size_t)M * 512 * 4);   // in_proj out (im2col bf16 aliases here)
    float* XM    = (float*)alloc((size_t)M * 256 * 4);
    float* XDBL  = (float*)alloc((size_t)M * 40 * 4);
    float* DELTA = (float*)alloc((size_t)M * 256 * 4);
    float* YV    = (float*)alloc((size_t)M * 256 * 4);
    bf16*  ABF   = (bf16*)alloc((size_t)M * 256 * 2);
    bf16*  WPACK = (bf16*)alloc((size_t)65536 * 2);

    // 1) front conv as im2col GEMM: (M, 448) @ (448, 128) + bias
    bf16* IM2 = (bf16*)XZ;
    im2col_conv<<<CDIV((long long)M * IN * 7, 256), 256, 0, stream>>>(x, IM2, B, IN, L, 7, 3);
    pack_b_bf16<<<CDIV(128 * 448, 256), 256, 0, stream>>>(conv_w, WPACK, 128, 448);
    launch_gemm(IM2, WPACK, conv_b, T0, M, 128, 448, stream);

    // 2) convln:  T1 = LN(T0)*w + b + T0
    ln_tokens<<<CDIV(M, 8), 256, 0, stream>>>(T0, nullptr, convln_w, convln_b, T0, T1, M);

    // 3) mamba1: T1 -> T2
    run_mamba(m1, T1, T2, XZ, XM, XDBL, DELTA, YV, ABF, WPACK, stream);

    // 4) ln1: T1 = LN(T2)
    ln_tokens<<<CDIV(M, 8), 256, 0, stream>>>(T2, nullptr, ln1_w, ln1_b, nullptr, T1, M);

    // 5) mamba2: T1 -> T0
    run_mamba(m2, T1, T0, XZ, XM, XDBL, DELTA, YV, ABF, WPACK, stream);

    // 6) ln2 with residual: T1 = LN(T0 + T2)
    ln_tokens<<<CDIV(M, 8), 256, 0, stream>>>(T0, T2, ln2_w, ln2_b, nullptr, T1, M);

    // 7) (B*L,128) -> (B,128,L)
    transpose_out<<<CDIV((long long)M * 128, 256), 256, 0, stream>>>(T1, out, B, L);
}
